// SelfAttention_90082644066489
// MI455X (gfx1250) — matrix operand order repaired and checked by reading, not compiled, not run on hardware
//
#include <hip/hip_runtime.h>
#include <hip/hip_bf16.h>
#include <math.h>

// Problem constants (match reference)
constexpr int B   = 32;
constexpr int C   = 256;
constexpr int L   = 1024;
constexpr int CQK = 32;   // C / 8

typedef __attribute__((ext_vector_type(16))) _Float16 v16h;
typedef __attribute__((ext_vector_type(8)))  _Float16 v8h;
typedef __attribute__((ext_vector_type(8)))  float    v8f;

// ---------------------------------------------------------------------------
// Kernel 1: fused q/k/v 1x1-conv projections.
//   x:[B,C,L] f32 -> qT:[B,L,32] f16, kT:[B,L,32] f16, vh:[B,C,L] f16
// qT/kT are stored position-major so WMMA A/B fragment loads are contiguous.
// One block handles (b, 32-position tile); x tile staged through LDS.
// ---------------------------------------------------------------------------
__global__ __launch_bounds__(256) void proj_qkv_kernel(
    const float* __restrict__ x,
    const float* __restrict__ Wq, const float* __restrict__ bq,
    const float* __restrict__ Wk, const float* __restrict__ bk,
    const float* __restrict__ Wv, const float* __restrict__ bv,
    _Float16* __restrict__ qT, _Float16* __restrict__ kT,
    _Float16* __restrict__ vh)
{
    __shared__ float sx[C][32];              // 32 KB
    const int b  = blockIdx.y;
    const int l0 = blockIdx.x * 32;
    const int tid = threadIdx.x;

    // Stage x[b, :, l0..l0+31] into LDS (coalesced over l).
    for (int i = tid; i < C * 32; i += 256) {
        const int c = i >> 5, l = i & 31;
        sx[c][l] = x[((size_t)b * C + c) * L + l0 + l];
    }
    __syncthreads();

    // 320 output channels (32 q + 32 k + 256 v) x 32 positions.
    for (int idx = tid; idx < 320 * 32; idx += 256) {
        const int o = idx >> 5;
        const int l = idx & 31;
        const float* w;
        float acc;
        if (o < 32)        { w = Wq + (size_t)o * C;        acc = bq[o];      }
        else if (o < 64)   { w = Wk + (size_t)(o - 32) * C; acc = bk[o - 32]; }
        else               { w = Wv + (size_t)(o - 64) * C; acc = bv[o - 64]; }
        #pragma unroll 8
        for (int c = 0; c < C; ++c)
            acc = fmaf(w[c], sx[c][l], acc);

        const int gl = l0 + l;
        if (o < 32)
            qT[((size_t)b * L + gl) * CQK + o] = (_Float16)acc;
        else if (o < 64)
            kT[((size_t)b * L + gl) * CQK + (o - 32)] = (_Float16)acc;
        else
            vh[((size_t)(b * C + (o - 64))) * L + gl] = (_Float16)acc;
    }
}

// wave32 reductions across the 16-lane half-group (lanes differing in bits 0..3)
__device__ __forceinline__ float hg_max(float v) {
    v = fmaxf(v, __shfl_xor(v, 1, 32));
    v = fmaxf(v, __shfl_xor(v, 2, 32));
    v = fmaxf(v, __shfl_xor(v, 4, 32));
    v = fmaxf(v, __shfl_xor(v, 8, 32));
    return v;
}
__device__ __forceinline__ float hg_sum(float v) {
    v += __shfl_xor(v, 1, 32);
    v += __shfl_xor(v, 2, 32);
    v += __shfl_xor(v, 4, 32);
    v += __shfl_xor(v, 8, 32);
    return v;
}

// ---------------------------------------------------------------------------
// Kernel 2: scores = qT x k (one v_wmma_f32_16x16x32_f16 per 16x16 tile,
// CQK==32 so the whole K-reduction is a single WMMA), softmax over j entirely
// in registers (8 waves each own a 16-row x 128-col stripe), then stream the
// attention out as f32 (required output) and f16 (operand for kernel 3).
// Block = 256 threads = 8 waves; grid = (L/16, B).
// ---------------------------------------------------------------------------
__global__ __launch_bounds__(256) void scores_softmax_kernel(
    const _Float16* __restrict__ qT, const _Float16* __restrict__ kT,
    float* __restrict__ attnf, _Float16* __restrict__ attnh)
{
    __shared__ float red[16][8];   // per-row, per-wave partials
    __shared__ float rowstat[16];  // row max, then row 1/sum

    const int b    = blockIdx.y;
    const int i0   = blockIdx.x * 16;
    const int lane = threadIdx.x & 31;
    const int wave = threadIdx.x >> 5;
    const int n    = lane & 15;    // fragment row/col within half-group
    const int half = lane >> 4;

    // --- A fragment: q rows i0..i0+15, K = 0..31 (ISA 16-bit A 16x32 layout:
    // lanes 0-15 hold K 0-7 & 16-23, lanes 16-31 hold K 8-15 & 24-31).
    const _Float16* ab = qT + ((size_t)(b * L + i0 + n)) * CQK;
    union { v16h v; v8h h[2]; } A;
    A.h[0] = *(const v8h*)(ab + half * 8);
    A.h[1] = *(const v8h*)(ab + half * 8 + 16);

    // --- 8 WMMA tiles per wave: columns j = (wave*8+t)*16 .. +15
    v8f accs[8];
    #pragma unroll
    for (int t = 0; t < 8; ++t) {
        const int j0 = (wave * 8 + t) * 16;
        // B fragment (32x16): lane n = column, lanes 0-15 K 0-7 & 16-23, 16-31 K 8-15 & 24-31
        const _Float16* bb = kT + ((size_t)(b * L + j0 + n)) * CQK + half * 8;
        union { v16h v; v8h h[2]; } Bu;
        Bu.h[0] = *(const v8h*)bb;
        Bu.h[1] = *(const v8h*)(bb + 16);
        const v16h Bf = Bu.v;
        v8f c = {};
        accs[t] = __builtin_amdgcn_wmma_f32_16x16x32_f16(
            false, A.v, false, Bf, (short)0, c, false, false);
    }

    // --- row max: lane-local over 8 tiles, then across half-group, then waves
    float pmax[8];
    #pragma unroll
    for (int r = 0; r < 8; ++r) {
        float m = accs[0][r];
        #pragma unroll
        for (int t = 1; t < 8; ++t) m = fmaxf(m, accs[t][r]);
        pmax[r] = hg_max(m);
    }
    if (n == 0) {
        #pragma unroll
        for (int r = 0; r < 8; ++r) red[r + 8 * half][wave] = pmax[r];
    }
    __syncthreads();
    if (threadIdx.x < 16) {
        float m = red[threadIdx.x][0];
        #pragma unroll
        for (int w = 1; w < 8; ++w) m = fmaxf(m, red[threadIdx.x][w]);
        rowstat[threadIdx.x] = m;
    }
    __syncthreads();

    // --- exp + row sum
    float psum[8];
    #pragma unroll
    for (int r = 0; r < 8; ++r) {
        const float mx = rowstat[r + 8 * half];
        float s = 0.0f;
        #pragma unroll
        for (int t = 0; t < 8; ++t) {
            const float e = __expf(accs[t][r] - mx);
            accs[t][r] = e;
            s += e;
        }
        psum[r] = hg_sum(s);
    }
    __syncthreads();   // everyone done reading rowstat (max) before overwrite
    if (n == 0) {
        #pragma unroll
        for (int r = 0; r < 8; ++r) red[r + 8 * half][wave] = psum[r];
    }
    __syncthreads();
    if (threadIdx.x < 16) {
        float s = red[threadIdx.x][0];
        #pragma unroll
        for (int w = 1; w < 8; ++w) s += red[threadIdx.x][w];
        rowstat[threadIdx.x] = 1.0f / s;
    }
    __syncthreads();

    // --- normalize and stream out (f32 required output + f16 shadow)
    #pragma unroll
    for (int r = 0; r < 8; ++r) {
        const int   m   = r + 8 * half;
        const float inv = rowstat[m];
        const size_t rowbase = ((size_t)b * L + (i0 + m)) * L;
        #pragma unroll
        for (int t = 0; t < 8; ++t) {
            const int j = (wave * 8 + t) * 16 + n;
            const float a = accs[t][r] * inv;
            attnf[rowbase + j] = a;
            attnh[rowbase + j] = (_Float16)a;
        }
    }
}

// ---------------------------------------------------------------------------
// Kernel 3: out[b,c,m] = gamma * sum_l vh[b,c,l]*att16[b,m,l] + x[b,c,m].
// One block per (b, 16-wide m stripe). The 16-row x 1024-K attention stripe
// (32 KB f16) is staged ONCE into LDS via global_load_async_to_lds_b128
// (ASYNCcnt async copy) and shared by all 8 waves, instead of each c-tile
// re-fetching it (16x read amplification). Each wave owns two c-tiles ->
// two independent WMMA accumulators sharing one B fragment (XDL pipelining).
// ---------------------------------------------------------------------------
constexpr int BPAD = 8;                       // halves of padding per LDS row
__global__ __launch_bounds__(256) void out_gemm_kernel(
    const _Float16* __restrict__ vh, const _Float16* __restrict__ attnh,
    const float* __restrict__ x, const float* __restrict__ gamma,
    float* __restrict__ out)
{
    __shared__ _Float16 sB[16][L + BPAD];     // ~33 KB attention stripe

    const int b    = blockIdx.y;
    const int m0   = blockIdx.x * 16;
    const int tid  = threadIdx.x;
    const int lane = tid & 31;
    const int wave = tid >> 5;
    const int n    = lane & 15;
    const int half = lane >> 4;

    // ---- async-stage attention rows m0..m0+15 (full K) into LDS ----
    // 8 passes x (16 rows x 16 chunks of 8 halves) = 16 x 1024 halves.
    {
        const int row = tid >> 4;             // 0..15
        const int chk = tid & 15;             // 0..15
        const size_t grow = ((size_t)b * L + m0 + row) * L;
        #pragma unroll
        for (int p = 0; p < 8; ++p) {
            const int koff = p * 128 + chk * 8;               // halves
            const unsigned lds = (unsigned)(size_t)&sB[row][koff];
            const unsigned long long ga =
                (unsigned long long)(attnh + grow + koff);
            asm volatile("global_load_async_to_lds_b128 %0, %1, off"
                         :: "v"(lds), "v"(ga) : "memory");
        }
        asm volatile("s_wait_asynccnt 0x0" ::: "memory");
    }
    __syncthreads();

    // ---- two c-tiles per wave, K=1024 in 32 WMMA steps ----
    const _Float16* arow0 = vh + ((size_t)(b * C + wave * 32 + n)) * L;
    const _Float16* arow1 = arow0 + (size_t)16 * L;

    v8f acc0 = {}, acc1 = {};
    for (int kk = 0; kk < L; kk += 32) {
        // B fragment from LDS: lane n = output column m, K 8h..8h+7 & 16+8h..16+8h+7
        union { v16h v; v8h h[2]; } Bu;
        Bu.h[0] = *(const v8h*)&sB[n][kk + half * 8];
        Bu.h[1] = *(const v8h*)&sB[n][kk + half * 8 + 16];

        union { v16h v; v8h h[2]; } A0, A1;
        A0.h[0] = *(const v8h*)(arow0 + kk + half * 8);
        A0.h[1] = *(const v8h*)(arow0 + kk + half * 8 + 16);
        A1.h[0] = *(const v8h*)(arow1 + kk + half * 8);
        A1.h[1] = *(const v8h*)(arow1 + kk + half * 8 + 16);
        if (kk + 32 < L)
            __builtin_prefetch(arow0 + kk + 32, 0, 0);        // global_prefetch

        acc0 = __builtin_amdgcn_wmma_f32_16x16x32_f16(
            false, A0.v, false, Bu.v, (short)0, acc0, false, false);
        acc1 = __builtin_amdgcn_wmma_f32_16x16x32_f16(
            false, A1.v, false, Bu.v, (short)0, acc1, false, false);
    }

    // ---- epilogue: out = gamma*acc + x ----
    const float g = gamma[0];
    const int m = m0 + n;                     // D: N = lane&15
    #pragma unroll
    for (int r = 0; r < 8; ++r) {
        const int c0 = wave * 32 + r + 8 * half;              // D: M = r+8*half
        const size_t o0 = ((size_t)(b * C + c0)) * L + m;
        out[o0] = g * acc0[r] + x[o0];
        const size_t o1 = o0 + (size_t)16 * L;
        out[o1] = g * acc1[r] + x[o1];
    }
}

// ---------------------------------------------------------------------------
// Host-side launcher
// ---------------------------------------------------------------------------
extern "C" void kernel_launch(void* const* d_in, const int* in_sizes, int n_in,
                              void* d_out, int out_size, void* d_ws, size_t ws_size,
                              hipStream_t stream) {
    const float* x     = (const float*)d_in[0];
    const float* Wq    = (const float*)d_in[1];
    const float* bq    = (const float*)d_in[2];
    const float* Wk    = (const float*)d_in[3];
    const float* bk    = (const float*)d_in[4];
    const float* Wv    = (const float*)d_in[5];
    const float* bv    = (const float*)d_in[6];
    const float* gamma = (const float*)d_in[7];

    float* out   = (float*)d_out;                      // [B,C,L] f32
    float* attnf = out + (size_t)B * C * L;            // [B,L,L] f32

    // workspace: f16 qT | kT | vh | att16  (~84 MB)
    _Float16* qT    = (_Float16*)d_ws;
    _Float16* kT    = qT + (size_t)B * L * CQK;
    _Float16* vh    = kT + (size_t)B * L * CQK;
    _Float16* att16 = vh + (size_t)B * C * L;

    dim3 g1(L / 32, B);
    proj_qkv_kernel<<<g1, 256, 0, stream>>>(x, Wq, bq, Wk, bk, Wv, bv, qT, kT, vh);

    dim3 g2(L / 16, B);
    scores_softmax_kernel<<<g2, 256, 0, stream>>>(qT, kT, attnf, att16);

    dim3 g3(L / 16, B);
    out_gemm_kernel<<<g3, 256, 0, stream>>>(vh, att16, x, gamma, out);
}
